// SRGNN_68436008895080
// MI455X (gfx1250) — matrix operand order repaired and checked by reading, not compiled, not run on hardware
//
#include <hip/hip_runtime.h>
#include <hip/hip_bf16.h>

// ---------------- problem constants (fixed by reference setup) ----------------
#define Hh   64
#define Dd   512
#define Bb   4096
#define NPS  40
#define SPS  50
#define EPS  60
#define Nn   (Bb * NPS)   // 163840 nodes
#define Tt   (Bb * SPS)   // 204800 tokens
#define Ee   (Bb * EPS)   // 245760 edges

typedef __attribute__((ext_vector_type(16))) _Float16 v16h;
typedef __attribute__((ext_vector_type(8)))  float    v8f;

// ---------------- weight downcast + transpose: WT[c*rows + r] = (f16)W[r*cols + c] ----
__global__ void k_castT(const float* __restrict__ W, _Float16* __restrict__ WT,
                        int rows, int cols) {
    int i = blockIdx.x * blockDim.x + threadIdx.x;
    if (i >= rows * cols) return;
    int r = i / cols, c = i % cols;
    WT[(size_t)c * rows + r] = (_Float16)W[i];
}

// ---------------- token -> global node index ---------------------------------------
__global__ void k_tok_node(const int* __restrict__ sid, int* __restrict__ tn) {
    int t = blockIdx.x * blockDim.x + threadIdx.x;
    if (t < Tt) tn[t] = (t / SPS) * NPS + sid[t];
}

// ---------------- SGConv: whole 2-hop propagation per session inside LDS -----------
// One 256-thread workgroup per session. LDS ping-pong buffers 2 x 40x512 f32 (160KB,
// legal on CDNA5's 320KB/WGP LDS). Thread t owns dims {2t, 2t+1} of EVERY node, so
// the sequential edge loop needs no atomics and no inter-thread sync.
__global__ __launch_bounds__(256)
void k_sgconv(const float* __restrict__ hidden, const int* __restrict__ ei,
              float* __restrict__ out) {
    extern __shared__ float smem[];
    float* buf0  = smem;                    // 40*512
    float* buf1  = buf0 + NPS * Dd;         // 40*512
    float* sdinv = buf1 + NPS * Dd;         // 40
    int*   esrc  = (int*)(sdinv + NPS);     // 60
    int*   edst  = esrc + EPS;              // 60

    const int s   = blockIdx.x;
    const int tid = threadIdx.x;

    // stage local edge list
    for (int e = tid; e < EPS; e += 256) {
        esrc[e] = ei[s * EPS + e]      - s * NPS;
        edst[e] = ei[Ee + s * EPS + e] - s * NPS;
    }
    // stage node features (coalesced b128)
    const float* gx = hidden + (size_t)s * NPS * Dd;
    for (int i = tid; i < NPS * Dd / 4; i += 256)
        ((float4*)buf0)[i] = ((const float4*)gx)[i];
    __syncthreads();

    // degrees: deg[n] = 1 (self loop) + #edges with dst==n
    if (tid < NPS) {
        float dg = 1.0f;
        for (int e = 0; e < EPS; ++e) dg += (edst[e] == tid) ? 1.0f : 0.0f;
        sdinv[tid] = __frsqrt_rn(dg);
    }
    __syncthreads();

    const int d0 = tid * 2;                 // this thread's private dims
    float* x = buf0;
    float* y = buf1;
#pragma unroll
    for (int hop = 0; hop < 2; ++hop) {
        // self loops initialize y
        for (int n = 0; n < NPS; ++n) {
            float nm = sdinv[n] * sdinv[n];
            float2 v = *(float2*)&x[n * Dd + d0];
            v.x *= nm; v.y *= nm;
            *(float2*)&y[n * Dd + d0] = v;
        }
        // edge accumulation (dims d0,d0+1 are thread-private -> race-free)
        for (int e = 0; e < EPS; ++e) {
            int a = esrc[e], b = edst[e];
            float nm = sdinv[a] * sdinv[b];
            float2 xv = *(float2*)&x[a * Dd + d0];
            float2 yv = *(float2*)&y[b * Dd + d0];
            yv.x += nm * xv.x; yv.y += nm * xv.y;
            *(float2*)&y[b * Dd + d0] = yv;
        }
        float* t = x; x = y; y = t;
    }
    // x now holds the 2-hop result; write back coalesced (thread-stride-2 floats)
    float* gout = out + (size_t)s * NPS * Dd;
    for (int n = 0; n < NPS; ++n)
        *(float2*)&gout[n * Dd + d0] = *(float2*)&x[n * Dd + d0];
}

// ---------------- generic WMMA GEMM with A-fragment reuse --------------------------
// C[M,Nout] = gather(A)[M,K](f32) * Bt^T + bias.  A row-major [*,K] f32 (row via idx
// if non-null), Bt f16 [Nout,K].  One wave computes a 16 x (16*NT) strip: the A
// fragment is loaded once per k-step and reused across NT independent WMMAs.
template<int NT>
__global__ __launch_bounds__(32)
void k_gemm_wmma(const float* __restrict__ A, const int* __restrict__ idx,
                 const _Float16* __restrict__ Bt, const float* __restrict__ bias,
                 float* __restrict__ C, int Kdim, int Nout) {
    const int bm   = blockIdx.x * 16;
    const int bn   = blockIdx.y * 16 * NT;
    const int lane = threadIdx.x;        // 0..31 (wave32)
    const int g    = lane >> 4;          // lane-group (K-half select)
    const int mr   = lane & 15;

    int arow = bm + mr;
    if (idx) arow = idx[arow];
    const float*    Arow = A  + (size_t)arow      * Kdim;
    const _Float16* Bcol = Bt + (size_t)(bn + mr) * Kdim;

    v8f acc[NT];
#pragma unroll
    for (int t = 0; t < NT; ++t) acc[t] = (v8f){};

    for (int k = 0; k < Kdim; k += 32) {
        // A fragment (16-bit A 16x32 layout): lanes 0-15 hold K 0-7 / 16-23,
        // lanes 16-31 hold K 8-15 / 24-31, for row M = lane%16.
        v16h a;
        const int ka = k + g * 8;
#pragma unroll
        for (int i = 0; i < 8; ++i) {
            a[i]     = (_Float16)Arow[ka + i];
            a[8 + i] = (_Float16)Arow[ka + 16 + i];
        }
        if (k + 64 < Kdim) __builtin_prefetch(Arow + k + 64, 0, 1);

#pragma unroll
        for (int t = 0; t < NT; ++t) {
            // B fragment: col N = lane%16; lanes 0-15 K 0-7 / 16-23, lanes 16-31 K 8-15 / 24-31
            const _Float16* bp = Bcol + (size_t)t * 16 * Kdim + k + g * 8;
            v16h b;
#pragma unroll
            for (int i = 0; i < 16; ++i) b[i] = bp[i < 8 ? i : i + 8];
            acc[t] = __builtin_amdgcn_wmma_f32_16x16x32_f16(
                /*neg_a=*/false, a, /*neg_b=*/false, b,
                /*c_mod=*/(short)0, acc[t], /*reuse_a=*/false, /*reuse_b=*/false);
        }
    }
    // D layout: VGPR r -> M = r + 8*(lane/16), N = lane%16
#pragma unroll
    for (int t = 0; t < NT; ++t) {
        const int col = bn + t * 16 + mr;
        const float bv = bias[col];
#pragma unroll
        for (int r = 0; r < 8; ++r) {
            int row = bm + r + 8 * g;
            C[(size_t)row * Nout + col] = acc[t][r] + bv;
        }
    }
}

// ---------------- v_n gather: last-token node embedding per session ----------------
__global__ void k_gather_vn(const float* __restrict__ xg, const int* __restrict__ sid,
                            float* __restrict__ vn) {
    int i = blockIdx.x * blockDim.x + threadIdx.x;      // over B*(D/4)
    if (i >= Bb * (Dd / 4)) return;
    int s = i / (Dd / 4), c = i % (Dd / 4);
    int t = s * SPS + (SPS - 1);
    int node = s * NPS + sid[t];
    ((float4*)(vn + (size_t)s * Dd))[c] =
        ((const float4*)(xg + (size_t)node * Dd))[c];
}

// ---------------- alpha[t] = sigmoid(vW1[s]+tW2[t]) . Wq + bq ----------------------
__global__ void k_alpha(const float* __restrict__ vW1, const float* __restrict__ tW2,
                        const float* __restrict__ Wq, const float* __restrict__ bq,
                        float* __restrict__ alpha) {
    int t = blockIdx.x * blockDim.x + threadIdx.x;
    if (t >= Tt) return;
    int s = t / SPS;
    const float* a = vW1 + (size_t)s * Hh;
    const float* b = tW2 + (size_t)t * Hh;
    float acc = bq[0];
#pragma unroll 8
    for (int h = 0; h < Hh; ++h) {
        float z = a[h] + b[h];
        acc += Wq[h] * (1.0f / (1.0f + __expf(-z)));
    }
    alpha[t] = acc;
}

// ---------------- s_g[s] = sum_t alpha[t] * xg[tok_node[t]] ------------------------
__global__ void k_sg(const float* __restrict__ alpha, const float* __restrict__ xg,
                     const int* __restrict__ tn, float* __restrict__ sg) {
    int i = blockIdx.x * blockDim.x + threadIdx.x;      // over B*D
    if (i >= Bb * Dd) return;
    int s = i / Dd, d = i % Dd;
    float acc = 0.0f;
#pragma unroll 5
    for (int j = 0; j < SPS; ++j) {
        int t = s * SPS + j;
        acc += alpha[t] * xg[(size_t)tn[t] * Dd + d];
    }
    sg[i] = acc;
}

// ---------------- cat = [v_n | s_g] ------------------------------------------------
__global__ void k_cat(const float* __restrict__ vn, const float* __restrict__ sg,
                      float* __restrict__ cat) {
    int i = blockIdx.x * blockDim.x + threadIdx.x;      // over B*(2D/4)
    if (i >= Bb * (2 * Dd / 4)) return;
    int s = i / (2 * Dd / 4), c = i % (2 * Dd / 4);
    float4 v = (c < Dd / 4) ? ((const float4*)(vn + (size_t)s * Dd))[c]
                            : ((const float4*)(sg + (size_t)s * Dd))[c - Dd / 4];
    ((float4*)(cat + (size_t)s * 2 * Dd))[c] = v;
}

// ==================================================================================
extern "C" void kernel_launch(void* const* d_in, const int* in_sizes, int n_in,
                              void* d_out, int out_size, void* d_ws, size_t ws_size,
                              hipStream_t stream) {
    (void)in_sizes; (void)n_in; (void)out_size; (void)ws_size;

    const float* hidden = (const float*)d_in[0];
    const float* Wg  = (const float*)d_in[1];
    const float* bg  = (const float*)d_in[2];
    const float* W1  = (const float*)d_in[3];
    const float* b1  = (const float*)d_in[4];
    const float* W2  = (const float*)d_in[5];
    const float* b2  = (const float*)d_in[6];
    const float* Wq  = (const float*)d_in[7];
    const float* bq  = (const float*)d_in[8];
    const float* W3  = (const float*)d_in[9];
    const float* b3  = (const float*)d_in[10];
    const int*   ei  = (const int*)d_in[11];
    const int*   sid = (const int*)d_in[13];
    float* out = (float*)d_out;

    // ---- workspace carve-up ----
    const size_t ND = (size_t)Nn * Dd;
    float*     h1   = (float*)d_ws;            // N*D  (xg = post-GEMM embeddings)
    float*     h2   = h1 + ND;                 // N*D  (SGConv result; region reused)
    _Float16*  WgT  = (_Float16*)(h2 + ND);    // 512*512
    _Float16*  W1T  = WgT + 512 * 512;         // 64*512
    _Float16*  W2T  = W1T + 64 * 512;          // 64*512
    _Float16*  W3T  = W2T + 64 * 512;          // 64*1024
    float*     vn   = (float*)(W3T + 64 * 1024);     // B*512
    float*     vW1  = vn + (size_t)Bb * Dd;          // B*64
    int*       tn   = (int*)(vW1 + (size_t)Bb * Hh); // T
    // aliased into h2 region (dead after the Wg GEMM):
    float* tW2   = h2;                               // T*64
    float* alpha = tW2 + (size_t)Tt * Hh;            // T
    float* sg    = alpha + Tt;                       // B*512
    float* cat   = sg + (size_t)Bb * Dd;             // B*1024

    const int TB = 256;

    // ---- weights -> f16, transposed [out][in] ----
    k_castT<<<(512 * 512 + TB - 1) / TB, TB, 0, stream>>>(Wg, WgT, 512, 512);
    k_castT<<<(512 * 64 + TB - 1) / TB, TB, 0, stream>>>(W1, W1T, 512, 64);
    k_castT<<<(512 * 64 + TB - 1) / TB, TB, 0, stream>>>(W2, W2T, 512, 64);
    k_castT<<<(1024 * 64 + TB - 1) / TB, TB, 0, stream>>>(W3, W3T, 1024, 64);
    k_tok_node<<<(Tt + TB - 1) / TB, TB, 0, stream>>>(sid, tn);

    // ---- SGConv: full 2-hop propagation per session in LDS (160KB + tables) ----
    const size_t sgconv_lds = (size_t)(2 * NPS * Dd + NPS) * sizeof(float)
                            + (size_t)(2 * EPS) * sizeof(int);
    k_sgconv<<<Bb, 256, sgconv_lds, stream>>>(hidden, ei, h2);

    // ---- xg = h2 @ Wg + bg  (WMMA GEMM, 16x128 strip per wave, writes h1) ----
    k_gemm_wmma<8><<<dim3(Nn / 16, Dd / 128), 32, 0, stream>>>(
        h2, nullptr, WgT, bg, h1, Dd, Dd);

    // ---- attention pooling ----
    k_gather_vn<<<(Bb * (Dd / 4) + TB - 1) / TB, TB, 0, stream>>>(h1, sid, vn);
    k_gemm_wmma<4><<<dim3(Bb / 16, 1), 32, 0, stream>>>(vn, nullptr, W1T, b1, vW1, Dd, Hh);
    k_gemm_wmma<4><<<dim3(Tt / 16, 1), 32, 0, stream>>>(h1, tn, W2T, b2, tW2, Dd, Hh);
    k_alpha<<<(Tt + TB - 1) / TB, TB, 0, stream>>>(vW1, tW2, Wq, bq, alpha);
    k_sg<<<(Bb * Dd + TB - 1) / TB, TB, 0, stream>>>(alpha, h1, tn, sg);
    k_cat<<<(Bb * (2 * Dd / 4) + TB - 1) / TB, TB, 0, stream>>>(vn, sg, cat);

    // ---- h_s = [v_n | s_g] @ W3 + b3 -> d_out [B, 64] ----
    k_gemm_wmma<4><<<dim3(Bb / 16, 1), 32, 0, stream>>>(cat, nullptr, W3T, b3, out, 2 * Dd, Hh);
}
